// GeoDynamicLayer_28097676050953
// MI455X (gfx1250) — matrix operand order repaired and checked by reading, not compiled, not run on hardware
//
#include <hip/hip_runtime.h>

typedef _Float16 v4h  __attribute__((ext_vector_type(4)));
typedef _Float16 v8h  __attribute__((ext_vector_type(8)));
typedef _Float16 v16h __attribute__((ext_vector_type(16)));
typedef float    v8f  __attribute__((ext_vector_type(8)));

#define IN_DIM 512
#define OUT_DIM 1024
#define NROWS 32768   // 8 * 4096

// ---------------------------------------------------------------- utilities
__global__ void k_zero(float* __restrict__ p, int n) {
    int i = blockIdx.x * blockDim.x + threadIdx.x;
    if (i < n) p[i] = 0.0f;
}

// column sums of x [32768, 512] -> zsum[512]
__global__ __launch_bounds__(512) void k_colsum(const float* __restrict__ x,
                                                float* __restrict__ zsum) {
    int c = threadIdx.x;                 // 512 threads = 512 columns
    long r0 = (long)blockIdx.x * 64;     // 512 blocks * 64 rows
    float s = 0.0f;
    for (int i = 0; i < 64; ++i) s += x[(r0 + i) * IN_DIM + c];
    atomicAdd(&zsum[c], s);
}

// 16x16 matmul helper on LDS (t = threadIdx.x, active for t < 256)
__device__ inline void mm16(const float* A, const float* B, float* C, int t) {
    if (t < 256) {
        int i = t >> 4, j = t & 15;
        float s = 0.0f;
#pragma unroll
        for (int k = 0; k < 16; ++k) s += A[i * 16 + k] * B[k * 16 + j];
        C[t] = s;
    }
}

// ------------------------------------------------- all small low-rank algebra
// Produces: Qg [512x16], M8g [16x16], Hg [8x512]
__global__ __launch_bounds__(512) void k_small(
        const float* __restrict__ zsum,
        const float* __restrict__ Wa, const float* __restrict__ La,
        const float* __restrict__ Ra, const float* __restrict__ Wb,
        const float* __restrict__ Rb,
        float* __restrict__ Qg, float* __restrict__ M8g, float* __restrict__ Hg) {
    __shared__ float zbar[512];
    __shared__ float ab[8], bb[8];
    __shared__ float Km[256], A2[256], A3[256], Rt[256], St[256];
    __shared__ float Mm[256], Tt[256], t1[256], t2m[256], Nt[256];
    __shared__ float RbP[128], Ff[128];

    const int t = threadIdx.x;
    // P1: pooled mean
    zbar[t] = zsum[t] * (1.0f / (float)NROWS);
    __syncthreads();
    // P2: abar, bbar (rank coefficients)
    if (t < 8) {
        float s = 0.0f;
        for (int c = 0; c < IN_DIM; ++c) s += zbar[c] * Wa[c * 8 + t];
        ab[t] = s;
    } else if (t < 16) {
        int r = t - 8;
        float s = 0.0f;
        for (int c = 0; c < IN_DIM; ++c) s += zbar[c] * Wb[c * 8 + r];
        bb[r] = s;
    }
    __syncthreads();
    // P3: K = Q^T P (16x16 core) and RbP = Rb^T P (8x16)
    if (t < 256) {
        int i = t >> 4, j = t & 15;
        float s = 0.0f;
        for (int c = 0; c < IN_DIM; ++c) {
            float qv = (i < 8) ? ab[i] * Ra[c * 8 + i] : -ab[i - 8] * La[c * 8 + (i - 8)];
            float pv = (j < 8) ? La[c * 8 + j] : Ra[c * 8 + (j - 8)];
            s += qv * pv;
        }
        Km[t] = s;
    } else if (t < 384) {
        int u = t - 256, r = u >> 4, j = u & 15;
        float s = 0.0f;
        for (int c = 0; c < IN_DIM; ++c) {
            float pv = (j < 8) ? La[c * 8 + j] : Ra[c * 8 + (j - 8)];
            s += Rb[c * 8 + r] * pv;
        }
        RbP[u] = s;
    }
    __syncthreads();
    // P4/P5: K^2, K^3
    mm16(Km, Km, A2, t); __syncthreads();
    mm16(A2, Km, A3, t); __syncthreads();
    // P6: per-step maps Rt = h I + h^2/2 K + h^3/6 K^2 + h^4/24 K^3 (tilde)
    //     St = h^2/2 I + h^3/6 K + h^4/24 K^2 (tilde)
    if (t < 256) {
        const float h = 0.125f;
        const float c2 = h * h * 0.5f;
        const float c3 = h * h * h * (1.0f / 6.0f);
        const float c4 = h * h * h * h * (1.0f / 24.0f);
        float id = ((t >> 4) == (t & 15)) ? 1.0f : 0.0f;
        float rt = h * id + c2 * Km[t] + c3 * A2[t] + c4 * A3[t];
        float st = c2 * id + c3 * Km[t] + c4 * A2[t];
        Rt[t] = rt; St[t] = st;
        Mm[t] = rt;   // M~_1
        Tt[t] = rt;   // T~ accumulates M~_1..M~_7
    }
    __syncthreads();
    // P7: compose R^n:  M~_{n+1} = M~_n + R~ + M~_n K R~   (n = 1..6 added to T)
    for (int it = 0; it < 6; ++it) {
        mm16(Mm, Km, t1, t);  __syncthreads();
        mm16(t1, Rt, t2m, t); __syncthreads();
        if (t < 256) {
            float nm = Mm[t] + Rt[t] + t2m[t];
            Mm[t] = nm;
            Tt[t] += nm;
        }
        __syncthreads();
    }
    // final composition -> M~_8 (not added to T)
    mm16(Mm, Km, t1, t);  __syncthreads();
    mm16(t1, Rt, t2m, t); __syncthreads();
    if (t < 256) Mm[t] = Mm[t] + Rt[t] + t2m[t];
    __syncthreads();
    // P9: N~ = h T~ + 8 S~ + S~ K T~
    mm16(St, Km, t1, t);  __syncthreads();
    mm16(t1, Tt, t2m, t); __syncthreads();
    if (t < 256) Nt[t] = 0.125f * Tt[t] + 8.0f * St[t] + t2m[t];
    __syncthreads();
    // P10: F = diag(b) RbP N~  (8x16)
    if (t < 128) {
        int r = t >> 4, j = t & 15;
        float s = 0.0f;
#pragma unroll
        for (int k = 0; k < 16; ++k) s += RbP[r * 16 + k] * Nt[k * 16 + j];
        Ff[t] = bb[r] * s;
    }
    __syncthreads();
    // P11: emit Q [512x16], H [8x512], M8 [16x16]
    {
        int c = t;
        float q[16];
#pragma unroll
        for (int j = 0; j < 8; ++j) {
            q[j]     = ab[j] * Ra[c * 8 + j];
            q[j + 8] = -ab[j] * La[c * 8 + j];
        }
#pragma unroll
        for (int j = 0; j < 16; ++j) Qg[c * 16 + j] = q[j];
#pragma unroll
        for (int r = 0; r < 8; ++r) {
            float s = bb[r] * Rb[c * 8 + r];
#pragma unroll
            for (int j = 0; j < 16; ++j) s += Ff[r * 16 + j] * q[j];
            Hg[r * IN_DIM + c] = s;
        }
    }
    if (t < 256) M8g[t] = Mm[t];
}

// ---------------------------------------------- V2 = (U0 @ P) @ M8  [1024x16]
__global__ __launch_bounds__(256) void k_V(const float* __restrict__ U0,
                                           const float* __restrict__ La,
                                           const float* __restrict__ Ra,
                                           const float* __restrict__ M8g,
                                           float* __restrict__ V2) {
    int e = blockIdx.x * 256 + threadIdx.x;
    int o = e >> 4, j = e & 15;
    float s = 0.0f;
    for (int c = 0; c < IN_DIM; ++c) {
        float pv = (j < 8) ? La[c * 8 + j] : Ra[c * 8 + (j - 8)];
        s += U0[o * IN_DIM + c] * pv;
    }
    __shared__ float Vl[256];
    Vl[threadIdx.x] = s;
    __syncthreads();
    int ol = threadIdx.x >> 4;
    float s2 = 0.0f;
#pragma unroll
    for (int k = 0; k < 16; ++k) s2 += Vl[ol * 16 + k] * M8g[k * 16 + j];
    V2[o * 16 + j] = s2;
}

// ------------------------------ W = U0 + V2 Q^T + [0; Lb H], stored as f16
__global__ __launch_bounds__(256) void k_buildW(
        const float* __restrict__ U0, const float* __restrict__ Lb,
        const float* __restrict__ Qg, const float* __restrict__ V2,
        const float* __restrict__ Hg, _Float16* __restrict__ Wh) {
    int e = blockIdx.x * 256 + threadIdx.x;
    int o = e >> 9;        // 0..1023
    int c = e & 511;
    float w = U0[o * IN_DIM + c];
    const float* v2 = &V2[o * 16];
    const float* qr = &Qg[c * 16];
#pragma unroll
    for (int j = 0; j < 16; ++j) w += v2[j] * qr[j];
    if (o >= IN_DIM) {
        int o2 = o - IN_DIM;
#pragma unroll
        for (int r = 0; r < 8; ++r) w += Lb[o2 * 8 + r] * Hg[r * IN_DIM + c];
    }
    Wh[o * IN_DIM + c] = (_Float16)w;
}

// ---------------------------------------------- big GEMM: out = x @ W^T
// block tile 128(M) x 128(N), K-step 32, 8 waves (2Mx4N), wave tile 64x32.
// Double-buffered LDS; x staged via regs (f32->f16 cvt), W staged with
// global_load_async_to_lds_b128 (ASYNCcnt), v_wmma_f32_16x16x32_f16.
#define LPITCH 40   // LDS row pitch in halves (pad vs bank conflicts)

__global__ __launch_bounds__(256) void k_gemm(const float* __restrict__ x,
                                              const _Float16* __restrict__ Wh,
                                              float* __restrict__ out) {
    __shared__ _Float16 sx[2][128 * LPITCH];
    __shared__ _Float16 sw[2][128 * LPITCH];

    const int tid  = threadIdx.x;
    const int lane = tid & 31;
    const int wave = tid >> 5;
    const int lhi  = lane >> 4;    // 0/1: lane group
    const int llo  = lane & 15;
    const int mOff = (wave & 1) * 64;
    const int nOff = (wave >> 1) * 32;
    const long rowBase = (long)blockIdx.x * 128;
    const int  colBase = blockIdx.y * 128;

    // x staging coords: thread -> (row r0 + 32*rr, 4 cols at c4*4)
    const int c4 = tid & 7, r0 = tid >> 3;
    // W staging coords: thread -> rows wo and wo+64, 8 halves at wc8
    const int wo = tid >> 2, wc8 = (tid & 3) * 8;

    v8f acc[4][2] = {};
    float4 g[4];

    auto issue_x = [&](int kk) {
#pragma unroll
        for (int rr = 0; rr < 4; ++rr)
            g[rr] = *(const float4*)&x[(rowBase + r0 + rr * 32) * IN_DIM + kk + c4 * 4];
    };
    auto issue_w_async = [&](int kk, int p) {
        unsigned lds0 = (unsigned)(unsigned long long)(uintptr_t)&sw[p][wo * LPITCH + wc8];
        unsigned lds1 = (unsigned)(unsigned long long)(uintptr_t)&sw[p][(wo + 64) * LPITCH + wc8];
        const _Float16* g0 = &Wh[(long)(colBase + wo) * IN_DIM + kk + wc8];
        const _Float16* g1 = &Wh[(long)(colBase + wo + 64) * IN_DIM + kk + wc8];
        asm volatile("global_load_async_to_lds_b128 %0, %1, off"
                     :: "v"(lds0), "v"(g0) : "memory");
        asm volatile("global_load_async_to_lds_b128 %0, %1, off"
                     :: "v"(lds1), "v"(g1) : "memory");
    };
    auto store_x = [&](int p) {
#pragma unroll
        for (int rr = 0; rr < 4; ++rr) {
            v4h hv = {(_Float16)g[rr].x, (_Float16)g[rr].y,
                      (_Float16)g[rr].z, (_Float16)g[rr].w};
            *(v4h*)&sx[p][(r0 + rr * 32) * LPITCH + c4 * 4] = hv;
        }
    };
    auto compute = [&](int p) {
        // Load ALL fragments first so the 12 ds_load_b128 issue back-to-back
        // and waits stagger across the WMMA stream.
        v16h B[2], A[4];
#pragma unroll
        for (int ni = 0; ni < 2; ++ni) {
            // B (32x16): lane L, half h -> n = L%16, k = (h<8?h:h+8) + 8*(L/16)
            const _Float16* q = &sw[p][(nOff + ni * 16 + llo) * LPITCH + lhi * 8];
            v8h lo = *(const v8h*)q;
            v8h hi = *(const v8h*)(q + 16);
            B[ni] = __builtin_shufflevector(lo, hi, 0, 1, 2, 3, 4, 5, 6, 7,
                                            8, 9, 10, 11, 12, 13, 14, 15);
        }
#pragma unroll
        for (int mi = 0; mi < 4; ++mi) {
            // A (16x32): lane L, half h -> M = L%16, k = (h<8?h:h+8) + 8*(L/16)
            const _Float16* q = &sx[p][(mOff + mi * 16 + llo) * LPITCH + lhi * 8];
            v8h lo = *(const v8h*)q;         // K = 8*lhi + 0..7
            v8h hi = *(const v8h*)(q + 16);  // K = 16 + 8*lhi + 0..7
            A[mi] = __builtin_shufflevector(lo, hi, 0, 1, 2, 3, 4, 5, 6, 7,
                                            8, 9, 10, 11, 12, 13, 14, 15);
        }
#pragma unroll
        for (int mi = 0; mi < 4; ++mi)
#pragma unroll
            for (int ni = 0; ni < 2; ++ni)
                acc[mi][ni] = __builtin_amdgcn_wmma_f32_16x16x32_f16(
                    false, A[mi], false, B[ni], (short)0, acc[mi][ni], false, false);
    };

    issue_x(0);
    issue_w_async(0, 0);
    int p = 0;
    // steady state: next tile's global loads overlap this tile's WMMAs
    for (int kk = 0; kk < IN_DIM - 32; kk += 32) {
        store_x(p);                                        // waits x loads, cvt, ds_store
        asm volatile("s_wait_asynccnt 0x0" ::: "memory");  // W tile p landed in LDS
        __syncthreads();                                   // dscnt flush + barrier
        issue_x(kk + 32);
        issue_w_async(kk + 32, p ^ 1);
        compute(p);
        p ^= 1;
    }
    // last tile (no new issues)
    store_x(p);
    asm volatile("s_wait_asynccnt 0x0" ::: "memory");
    __syncthreads();
    compute(p);

    // epilogue: C/D layout lane L, vgpr v -> N = L%16, M = v + 8*(L/16)
#pragma unroll
    for (int mi = 0; mi < 4; ++mi)
#pragma unroll
        for (int ni = 0; ni < 2; ++ni) {
            int col = colBase + nOff + ni * 16 + llo;
#pragma unroll
            for (int v = 0; v < 8; ++v) {
                long row = rowBase + mOff + mi * 16 + v + 8 * lhi;
                out[row * OUT_DIM + col] = acc[mi][ni][v];
            }
        }
}

// --------------------------------------------------------------------- launch
extern "C" void kernel_launch(void* const* d_in, const int* in_sizes, int n_in,
                              void* d_out, int out_size, void* d_ws, size_t ws_size,
                              hipStream_t stream) {
    const float* x  = (const float*)d_in[0];
    const float* U0 = (const float*)d_in[1];
    const float* Wa = (const float*)d_in[2];
    const float* La = (const float*)d_in[3];
    const float* Ra = (const float*)d_in[4];
    const float* Wb = (const float*)d_in[5];
    const float* Lb = (const float*)d_in[6];
    const float* Rb = (const float*)d_in[7];
    float* out = (float*)d_out;

    float* fws  = (float*)d_ws;
    float* zsum = fws;                 // 512
    float* Qg   = zsum + 512;          // 512*16
    float* M8g  = Qg + 512 * 16;       // 256
    float* Hg   = M8g + 256;           // 8*512
    float* V2   = Hg + 8 * 512;        // 1024*16
    _Float16* Wh = (_Float16*)(V2 + 1024 * 16);  // 1024*512 f16

    k_zero<<<1, 512, 0, stream>>>(zsum, 512);
    k_colsum<<<512, 512, 0, stream>>>(x, zsum);
    k_small<<<1, 512, 0, stream>>>(zsum, Wa, La, Ra, Wb, Rb, Qg, M8g, Hg);
    k_V<<<64, 256, 0, stream>>>(U0, La, Ra, M8g, V2);
    k_buildW<<<2048, 256, 0, stream>>>(U0, Lb, Qg, V2, Hg, Wh);
    dim3 grid(NROWS / 128, OUT_DIM / 128);
    k_gemm<<<grid, 256, 0, stream>>>(x, Wh, out);
}
